// MultiHeadAttention_15582141350198
// MI455X (gfx1250) — matrix operand order repaired and checked by reading, not compiled, not run on hardware
//
#include <hip/hip_runtime.h>

typedef __attribute__((ext_vector_type(16))) __bf16 v16bf;
typedef __attribute__((ext_vector_type(8)))  float  v8f;
typedef int v4i_vec __attribute__((vector_size(16)));
typedef __attribute__((address_space(1))) v4i_vec* gptr_v4i;
typedef __attribute__((address_space(3))) v4i_vec* lptr_v4i;

#if defined(__has_builtin)
#if __has_builtin(__builtin_amdgcn_global_load_async_to_lds_b128)
#define HAVE_ASYNC_LDS 1
#endif
#endif
#ifndef HAVE_ASYNC_LDS
#define HAVE_ASYNC_LDS 0
#endif

union FragU {
  v16bf v;
  uint4 q[2];
};

__device__ __forceinline__ unsigned short f2bf(float f) {
  unsigned int u = __float_as_uint(f);
  unsigned int r = u + 0x7FFFu + ((u >> 16) & 1u);  // round-to-nearest-even
  return (unsigned short)(r >> 16);
}

// A fragment: 16x32 bf16 tile from row-major src (leading dim ld, elements).
// lane<16 : row = `row`, K = {k0+0..7} U {k0+16..23}
// lane>=16: row = `row`, K = {k0+8..15} U {k0+24..31}
__device__ __forceinline__ v16bf load_a_frag(const unsigned short* base, int ld,
                                             int row, int k0, int half) {
  FragU f;
  const unsigned short* p = base + (size_t)row * ld + k0 + half * 8;
  f.q[0] = *(const uint4*)(p);
  f.q[1] = *(const uint4*)(p + 16);
  return f.v;
}

// B fragment: 32x16 bf16 tile; column n of B = row `row` of W[N,K] row-major.
// lane<16 : K = {k0+0..7} U {k0+16..23}; lane>=16: K = {k0+8..15} U {k0+24..31}
__device__ __forceinline__ v16bf load_b_frag(const unsigned short* base, int ld,
                                             int row, int k0, int half) {
  FragU f;
  const unsigned short* p = base + (size_t)row * ld + k0 + half * 8;
  f.q[0] = *(const uint4*)(p);
  f.q[1] = *(const uint4*)(p + 16);
  return f.v;
}

__device__ __forceinline__ void wait_async_zero() {
#if HAVE_ASYNC_LDS
#if __has_builtin(__builtin_amdgcn_s_wait_asynccnt)
  __builtin_amdgcn_s_wait_asynccnt(0);
#else
  asm volatile("s_wait_asynccnt 0" ::: "memory");
#endif
#endif
}

// Stage one 128x32 bf16 k-slice (8KB) into LDS; 256 threads x 2 x 16B segs.
__device__ __forceinline__ void stage_tile(const unsigned short* __restrict__ gsrc,
                                           int gld, int grow0, int gcol0,
                                           unsigned short* lbase, int tid) {
#pragma unroll
  for (int i = 0; i < 2; ++i) {
    int seg = tid + i * 256;       // 0..511 16-byte segments
    int row = seg >> 2;            // 0..127
    int cs = (seg & 3) * 8;        // element column start
    const unsigned short* g = gsrc + (size_t)(grow0 + row) * gld + gcol0 + cs;
    unsigned short* l = lbase + row * 32 + cs;
#if HAVE_ASYNC_LDS
    __builtin_amdgcn_global_load_async_to_lds_b128((gptr_v4i)g, (lptr_v4i)l, 0, 0);
#else
    *(uint4*)l = *(const uint4*)g;
#endif
  }
}

__global__ void f32_to_bf16_kernel(const float* __restrict__ in,
                                   unsigned short* __restrict__ out, int n) {
  int i = blockIdx.x * blockDim.x + threadIdx.x;
  if (i < n) out[i] = f2bf(in[i]);
}

// Vh[b,s,h,d] (bf16) -> Vt[b,h,d,s] (bf16)
__global__ void transpose_v_kernel(const unsigned short* __restrict__ Vh,
                                   unsigned short* __restrict__ Vt) {
  int idx = blockIdx.x * blockDim.x + threadIdx.x;
  if (idx >= 2 * 2048 * 1024) return;
  int d = idx & 63;
  int h = (idx >> 6) & 15;
  int s = (idx >> 10) & 2047;
  int b = idx >> 21;
  Vt[(((size_t)(b * 16 + h) * 64 + d) * 2048) + s] = Vh[idx];
}

// C[M,N] = (A[M,K] @ W[N,K]^T + bias[N]) * scale
// Block: 256 threads = 8 waves (4x2), block tile 128x128, LDS double-buffered
// 128x32 k-slices filled via async global->LDS copies (ASYNCcnt).
__global__ void __launch_bounds__(256)
gemm_bf16_wmma(const unsigned short* __restrict__ A,
               const unsigned short* __restrict__ W,
               const float* __restrict__ bias,
               float* __restrict__ outF,          // fp32 output (or null)
               unsigned short* __restrict__ outH, // bf16 output (or null)
               int M, int N, int K, float scale) {
  __shared__ __align__(16) unsigned short tA[2][128 * 32];
  __shared__ __align__(16) unsigned short tW[2][128 * 32];

  int tid = threadIdx.x;
  int wv = tid >> 5;
  int lane = tid & 31, half = lane >> 4, mrow = lane & 15;
  int nblk = N >> 7;
  int mb = blockIdx.x / nblk, nb = blockIdx.x - mb * nblk;
  int mBase = mb * 128, nBase = nb * 128;
  int wm = wv & 3, wn = wv >> 2;  // 4 waves in M, 2 in N
  int wrow = wm * 32, wcol = wn * 64;

  v8f acc[2][4] = {};

  stage_tile(A, K, mBase, 0, &tA[0][0], tid);
  stage_tile(W, K, nBase, 0, &tW[0][0], tid);

  int ksteps = K >> 5;
  for (int kt = 0; kt < ksteps; ++kt) {
    int cur = kt & 1;
    wait_async_zero();   // my async fills of buffer `cur` have landed in LDS
    __syncthreads();     // everyone's fills done; buffer cur^1 free to restage
    if (kt + 1 < ksteps) {
      stage_tile(A, K, mBase, (kt + 1) * 32, &tA[cur ^ 1][0], tid);
      stage_tile(W, K, nBase, (kt + 1) * 32, &tW[cur ^ 1][0], tid);
    }
    v16bf af[2], wf[4];
#pragma unroll
    for (int i = 0; i < 2; ++i)
      af[i] = load_a_frag(&tA[cur][0], 32, wrow + i * 16 + mrow, 0, half);
#pragma unroll
    for (int j = 0; j < 4; ++j)
      wf[j] = load_b_frag(&tW[cur][0], 32, wcol + j * 16 + mrow, 0, half);
#pragma unroll
    for (int i = 0; i < 2; ++i)
#pragma unroll
      for (int j = 0; j < 4; ++j)
        acc[i][j] = __builtin_amdgcn_wmma_f32_16x16x32_bf16(
            false, af[i], false, wf[j], (short)0, acc[i][j], false, false);
  }

#pragma unroll
  for (int i = 0; i < 2; ++i) {
    int grow0 = mBase + wrow + i * 16;
#pragma unroll
    for (int j = 0; j < 4; ++j) {
      int col = nBase + wcol + j * 16 + mrow;
      float bv = bias[col];
#pragma unroll
      for (int r = 0; r < 8; ++r) {
        float v = (acc[i][j][r] + bv) * scale;
        size_t idx = (size_t)(grow0 + r + 8 * half) * N + col;
        if (outF) outF[idx] = v;
        else      outH[idx] = f2bf(v);
      }
    }
  }
}

// Causal flash attention. Q pre-scaled by 1/sqrt(HD). One wave = 16 query rows.
__global__ void __launch_bounds__(256)
attn_wmma(const unsigned short* __restrict__ Qh,   // [B,S,H*HD] bf16 (scaled)
          const unsigned short* __restrict__ Kh,   // [B,S,H*HD] bf16
          const unsigned short* __restrict__ Vth,  // [B,H,HD,S] bf16
          unsigned short* __restrict__ Oh) {       // [B,S,H*HD] bf16
  constexpr int S = 2048, H = 16, HD = 64, D = 1024;
  __shared__ __align__(16) unsigned short ldsP[8][16 * 32];

  int wv = threadIdx.x >> 5;
  int gw = blockIdx.x * (blockDim.x >> 5) + wv;
  int qt = gw & (S / 16 - 1);
  int bh = gw >> 7;
  int b = bh >> 4;
  int h = bh & 15;
  int qbase = qt * 16;

  int lane = threadIdx.x & 31, half = lane >> 4, mrow = lane & 15;

  const unsigned short* Qbase  = Qh + ((size_t)(b * S + qbase) * D + h * HD);
  const unsigned short* Kbase0 = Kh + ((size_t)b * S * D + h * HD);
  const unsigned short* Vbase  = Vth + ((size_t)(b * H + h) * HD) * S;

  v16bf aq[2];
#pragma unroll
  for (int kk = 0; kk < 2; ++kk)
    aq[kk] = load_a_frag(Qbase, D, mrow, kk * 32, half);

  v8f o[4] = {};
  float mi[8], li[8];
#pragma unroll
  for (int r = 0; r < 8; ++r) { mi[r] = -3.0e38f; li[r] = 0.0f; }

  unsigned short* P = &ldsP[wv][0];
  int qmax = qbase + 15;

  for (int t0 = 0; t0 <= qmax; t0 += 32) {
    v8f s[2];
#pragma unroll
    for (int j = 0; j < 2; ++j) {
      int tb = t0 + 16 * j;
      v8f sc = {};
      const unsigned short* Kb = Kbase0 + (size_t)tb * D;
#pragma unroll
      for (int kk = 0; kk < 2; ++kk) {
        v16bf bk = load_b_frag(Kb, D, mrow, kk * 32, half);
        sc = __builtin_amdgcn_wmma_f32_16x16x32_bf16(
            false, aq[kk], false, bk, (short)0, sc, false, false);
      }
      int ki = tb + mrow;
#pragma unroll
      for (int r = 0; r < 8; ++r) {
        int qi = qbase + r + 8 * half;
        if (ki > qi) sc[r] = -3.0e38f;  // causal mask
      }
      s[j] = sc;
    }
    // Online softmax; each row lives in one 16-lane half -> xor masks 1,2,4,8.
#pragma unroll
    for (int r = 0; r < 8; ++r) {
      float rm = fmaxf(s[0][r], s[1][r]);
#pragma unroll
      for (int msk = 1; msk < 16; msk <<= 1)
        rm = fmaxf(rm, __shfl_xor(rm, msk, 32));
      float mnew = fmaxf(mi[r], rm);
      float alpha = __expf(mi[r] - mnew);
      float p0 = __expf(s[0][r] - mnew);
      float p1 = __expf(s[1][r] - mnew);
      float rs = p0 + p1;
#pragma unroll
      for (int msk = 1; msk < 16; msk <<= 1)
        rs += __shfl_xor(rs, msk, 32);
      li[r] = li[r] * alpha + rs;
      mi[r] = mnew;
#pragma unroll
      for (int t = 0; t < 4; ++t) o[t][r] *= alpha;
      int m = r + 8 * half;
      P[m * 32 + mrow]      = f2bf(p0);  // C-layout -> LDS (16x32 tile)
      P[m * 32 + 16 + mrow] = f2bf(p1);
    }
    asm volatile("s_wait_dscnt 0" ::: "memory");
    FragU pf;  // re-read P tile as an A fragment (row = mrow)
    {
      const unsigned short* lp = P + mrow * 32;
      pf.q[0] = *(const uint4*)(lp + half * 8);
      pf.q[1] = *(const uint4*)(lp + 16 + half * 8);
    }
#pragma unroll
    for (int t = 0; t < 4; ++t) {
      v16bf bvf = load_b_frag(Vbase, S, t * 16 + mrow, t0, half);
      o[t] = __builtin_amdgcn_wmma_f32_16x16x32_bf16(
          false, pf.v, false, bvf, (short)0, o[t], false, false);
    }
  }
#pragma unroll
  for (int r = 0; r < 8; ++r) {
    float inv = 1.0f / li[r];
    int m = r + 8 * half;
    size_t rowoff = (size_t)(b * S + qbase + m) * D + h * HD;
#pragma unroll
    for (int t = 0; t < 4; ++t)
      Oh[rowoff + t * 16 + mrow] = f2bf(o[t][r] * inv);
  }
}

extern "C" void kernel_launch(void* const* d_in, const int* in_sizes, int n_in,
                              void* d_out, int out_size, void* d_ws, size_t ws_size,
                              hipStream_t stream) {
  (void)in_sizes; (void)n_in; (void)out_size; (void)ws_size;
  const float* X  = (const float*)d_in[0];
  const float* Wq = (const float*)d_in[1];
  const float* bq = (const float*)d_in[2];
  const float* Wk = (const float*)d_in[3];
  const float* bk = (const float*)d_in[4];
  const float* Wv = (const float*)d_in[5];
  const float* bv = (const float*)d_in[6];
  const float* Wo = (const float*)d_in[7];
  const float* bo = (const float*)d_in[8];

  const int Bb = 2, S = 2048, D = 1024, M = Bb * S;
  const size_t Xn = (size_t)M * D;  // 4,194,304
  const size_t Wn = (size_t)D * D;  // 1,048,576

  unsigned short* Xh  = (unsigned short*)d_ws;
  unsigned short* Wqh = Xh  + Xn;
  unsigned short* Wkh = Wqh + Wn;
  unsigned short* Wvh = Wkh + Wn;
  unsigned short* Woh = Wvh + Wn;
  unsigned short* Qh  = Woh + Wn;
  unsigned short* Kh  = Qh  + Xn;
  unsigned short* Vh  = Kh  + Xn;
  unsigned short* Vth = Vh  + Xn;
  unsigned short* Oh  = Vth + Xn;   // total ~56 MB of workspace

  f32_to_bf16_kernel<<<(int)((Xn + 255) / 256), 256, 0, stream>>>(X, Xh, (int)Xn);
  f32_to_bf16_kernel<<<(int)((Wn + 255) / 256), 256, 0, stream>>>(Wq, Wqh, (int)Wn);
  f32_to_bf16_kernel<<<(int)((Wn + 255) / 256), 256, 0, stream>>>(Wk, Wkh, (int)Wn);
  f32_to_bf16_kernel<<<(int)((Wn + 255) / 256), 256, 0, stream>>>(Wv, Wvh, (int)Wn);
  f32_to_bf16_kernel<<<(int)((Wn + 255) / 256), 256, 0, stream>>>(Wo, Woh, (int)Wn);

  // (M/128)*(N/128) = 256 blocks, 128x128 tile per block.
  int gblk = (M / 128) * (D / 128);
  gemm_bf16_wmma<<<gblk, 256, 0, stream>>>(Xh, Wqh, bq, nullptr, Qh, M, D, D, 0.125f);
  gemm_bf16_wmma<<<gblk, 256, 0, stream>>>(Xh, Wkh, bk, nullptr, Kh, M, D, D, 1.0f);
  gemm_bf16_wmma<<<gblk, 256, 0, stream>>>(Xh, Wvh, bv, nullptr, Vh, M, D, D, 1.0f);

  transpose_v_kernel<<<(int)((Xn + 255) / 256), 256, 0, stream>>>(Vh, Vth);

  // B*H*(S/16) = 4096 waves -> 512 blocks.
  attn_wmma<<<512, 256, 0, stream>>>(Qh, Kh, Vth, Oh);

  gemm_bf16_wmma<<<gblk, 256, 0, stream>>>(Oh, Woh, bo, (float*)d_out, nullptr, M, D, D, 1.0f);
}
